// _MockAttention_79250736546487
// MI455X (gfx1250) — matrix operand order repaired and checked by reading, not compiled, not run on hardware
//
#include <hip/hip_runtime.h>

// ---------------------------------------------------------------------------
// Types for CDNA5 WMMA / TDM
// ---------------------------------------------------------------------------
typedef __attribute__((ext_vector_type(16))) __bf16 v16bf;
typedef __attribute__((ext_vector_type(8)))  float  v8f;
typedef __attribute__((ext_vector_type(4)))  unsigned int u32x4;
typedef __attribute__((ext_vector_type(4)))  int    i32x4;
typedef __attribute__((ext_vector_type(8)))  int    i32x8;

#define BM 128
#define BN 128
#define BK 32
#define LDSS 40            // bf16 elements per LDS row (32 data + 8 pad = 80B, 16B-aligned)

// ---------------------------------------------------------------------------
// f32 -> bf16 (round to nearest even), elementwise
// ---------------------------------------------------------------------------
__device__ inline __bf16 f2bf(float f) {
    union { float f; unsigned u; } x; x.f = f;
    unsigned r = x.u + 0x7FFFu + ((x.u >> 16) & 1u);
    union { unsigned short s; __bf16 b; } y;
    y.s = (unsigned short)(r >> 16);
    return y.b;
}

__global__ __launch_bounds__(256) void cvt_f32_bf16(const float* __restrict__ in,
                                                    __bf16* __restrict__ out, int n) {
    int i = blockIdx.x * 256 + threadIdx.x;
    if (i < n) out[i] = f2bf(in[i]);
}

// ---------------------------------------------------------------------------
// Tensor Data Mover: DMA one [128 rows x 32 bf16] tile, row stride K elements,
// from global memory into LDS with 4-DWORD padding every 16 DWORDs
// (=> 40-bf16 LDS row stride matching LDSS). D# layout per CDNA5 ISA §8.3/8.4.
// ---------------------------------------------------------------------------
__device__ inline void tdm_load_tile(unsigned long long gaddr, unsigned lds_addr, int K) {
    u32x4 g0;
    g0[0] = 1u;                                                // count=1, user desc
    g0[1] = lds_addr;                                          // LDS dest (bytes)
    g0[2] = (unsigned)(gaddr & 0xFFFFFFFFull);                 // global_addr[31:0]
    g0[3] = ((unsigned)(gaddr >> 32) & 0x01FFFFFFu)            // global_addr[56:32]
            | 0x80000000u;                                     // type=2 ("image")
    i32x8 g1;
    // data_size=1 (2B) <<16 | pad_enable<<20 | pad_interval=3 (16 DW) <<22
    // | pad_amount=3 (4 DW) <<25
    g1[0] = (int)((1u << 16) | (1u << 20) | (3u << 22) | (3u << 25));
    g1[1] = (int)(((unsigned)K & 0xFFFFu) << 16);              // tensor_dim0[15:0]
    g1[2] = (int)(((unsigned)K >> 16) | (128u << 16));         // dim0[31:16] | tensor_dim1=128
    g1[3] = (int)(32u << 16);                                  // tile_dim0 = 32 elems
    g1[4] = 128;                                               // tile_dim1 = 128 rows
    g1[5] = K;                                                 // tensor_dim0_stride[31:0]
    g1[6] = 0;
    g1[7] = 0;
    i32x4 z4 = {};
#if __clang_major__ >= 23
    i32x8 z8 = {};
    __builtin_amdgcn_tensor_load_to_lds(g0, g1, z4, z4, z8, 0);
#else
    __builtin_amdgcn_tensor_load_to_lds(g0, g1, z4, z4, 0);
#endif
}

// ---------------------------------------------------------------------------
// Y[M,N] = A[M,K] @ W[N,K]^T   (A, W bf16 row-major K-contiguous; Y f32)
// Block: 256 threads = 8 waves; 128x128 block tile; K-step 32.
// Staging: TDM double-buffered global->LDS (no VGPR round-trip),
// synchronized with TENSORcnt + workgroup barriers.
// Wave (wm 0..3, wn 0..1) computes rows [wm*32,+32) x cols [wn*64,+64)
// as a 2x4 grid of v_wmma_f32_16x16x32_bf16 tiles.
// ---------------------------------------------------------------------------
__global__ __launch_bounds__(256) void gemm_bf16_wmma(
    const __bf16* __restrict__ A, const __bf16* __restrict__ W,
    float* __restrict__ Y, int M, int N, int K)
{
    __shared__ __bf16 As[2][BM * LDSS];
    __shared__ __bf16 Ws[2][BN * LDSS];

    const int t    = threadIdx.x;
    const int lane = t & 31;
    const int wave = t >> 5;
    const int wm   = wave & 3;        // 4 row groups of 32
    const int wn   = wave >> 2;       // 2 col groups of 64
    const int bm   = blockIdx.y * BM;
    const int bn   = blockIdx.x * BN;

    const unsigned long long abase =
        (unsigned long long)(const void*)A + ((unsigned long long)bm * (unsigned)K) * 2ull;
    const unsigned long long wbase =
        (unsigned long long)(const void*)W + ((unsigned long long)bn * (unsigned)K) * 2ull;
    const unsigned ldsA[2] = { (unsigned)(size_t)&As[0][0], (unsigned)(size_t)&As[1][0] };
    const unsigned ldsW[2] = { (unsigned)(size_t)&Ws[0][0], (unsigned)(size_t)&Ws[1][0] };

    v8f acc[2][4];
    const v8f vzero = {};
#pragma unroll
    for (int mi = 0; mi < 2; ++mi)
#pragma unroll
        for (int ni = 0; ni < 4; ++ni) acc[mi][ni] = vzero;

    // Fragment lane decomposition (ISA 7.12.2, 16-bit layouts, wave32):
    //  A 16x32: lane<16 -> row=lane, K = {koff..koff+7, koff+16..koff+23}, koff=(lane>>4)*8
    //  B 32x16 (= W row n, K-contiguous): col=lane&15, K = {koff..koff+7, koff+16..koff+23}
    const int r    = lane & 15;
    const int koff = (lane >> 4) * 8;
    const int kb16 = (lane >> 4) * 16;

    const int nt = K / BK;

    // prologue: DMA tile 0 into buffer 0 (one TDM issue per block, from wave 0)
    if (wave == 0) {
        tdm_load_tile(abase, ldsA[0], K);
        tdm_load_tile(wbase, ldsW[0], K);
    }

    for (int it = 0; it < nt; ++it) {
        const int buf = it & 1;
        if (wave == 0) __builtin_amdgcn_s_wait_tensorcnt(0);   // tile `it` landed
        __syncthreads();                                       // visible to all waves

        // overlap: DMA tile it+1 into the other buffer while we compute tile it
        if (wave == 0 && it + 1 < nt) {
            const unsigned long long koffb = (unsigned long long)(it + 1) * BK * 2ull;
            tdm_load_tile(abase + koffb, ldsA[buf ^ 1], K);
            tdm_load_tile(wbase + koffb, ldsW[buf ^ 1], K);
        }

        const __bf16* as = As[buf];
        const __bf16* ws = Ws[buf];

        v16bf afr[2], bfr[4];
#pragma unroll
        for (int mi = 0; mi < 2; ++mi) {
            const int arow = wm * 32 + mi * 16 + r;
            union { uint4 u[2]; v16bf v; } U;
            U.u[0] = *(const uint4*)(&as[arow * LDSS + koff]);        // K koff..koff+7
            U.u[1] = *(const uint4*)(&as[arow * LDSS + koff + 16]);   // K koff+16..koff+23
            afr[mi] = U.v;
        }
#pragma unroll
        for (int ni = 0; ni < 4; ++ni) {
            const int wrow = wn * 64 + ni * 16 + r;                   // output column
            union { uint4 u[2]; v16bf v; } U;
            U.u[0] = *(const uint4*)(&ws[wrow * LDSS + koff]);        // K koff..koff+7
            U.u[1] = *(const uint4*)(&ws[wrow * LDSS + koff + 16]);   // K koff+16..koff+23
            bfr[ni] = U.v;
        }

#pragma unroll
        for (int mi = 0; mi < 2; ++mi)
#pragma unroll
            for (int ni = 0; ni < 4; ++ni)
                acc[mi][ni] = __builtin_amdgcn_wmma_f32_16x16x32_bf16(
                    false, afr[mi], false, bfr[ni],
                    (short)0, acc[mi][ni], false, false);

        __syncthreads();   // tile `it` fully consumed before buffer is re-targeted
    }

    // ---- epilogue: C/D layout -> lane (row = (lane>>4)*8 + v, col = lane&15)
    const int rrow = (lane >> 4) * 8;
    const int col  = lane & 15;
#pragma unroll
    for (int mi = 0; mi < 2; ++mi)
#pragma unroll
        for (int ni = 0; ni < 4; ++ni)
#pragma unroll
            for (int v = 0; v < 8; ++v) {
                const size_t grow = (size_t)(bm + wm * 32 + mi * 16 + rrow + v);
                const size_t gcol = (size_t)(bn + wn * 64 + ni * 16 + col);
                Y[grow * N + gcol] = acc[mi][ni][v];
            }
}

// ---------------------------------------------------------------------------
// StreamingLLM attention: <=12 keys per query (4 sinks + 8-wide local window).
// One wave32 per (b, h, q); lane owns 4 of 128 head dims.
// Q:[B*S,2048] f32, K/V:[B*S,1024] f32, O:[B*S,2048] f32.
// ---------------------------------------------------------------------------
#define AS_B 2
#define AS_S 2048
#define AS_HQ 16
#define AS_D 128

__global__ __launch_bounds__(256) void attn_stream(
    const float* __restrict__ Q, const float* __restrict__ Kp,
    const float* __restrict__ Vp, float* __restrict__ O)
{
    const int gtid = blockIdx.x * 256 + threadIdx.x;
    const int w    = gtid >> 5;
    const int lane = gtid & 31;
    if (w >= AS_B * AS_HQ * AS_S) return;

    const int b   = w / (AS_HQ * AS_S);
    const int rem = w - b * (AS_HQ * AS_S);
    const int h   = rem / AS_S;
    const int i   = rem - h * AS_S;

    const size_t qoff  = ((size_t)(b * AS_S + i) * 2048) + h * AS_D + lane * 4;
    const float4 qv    = *(const float4*)(Q + qoff);
    const int    kvoff = (h >> 1) * AS_D + lane * 4;     // GQA: 2 q-heads per kv-head
    const float  scale = 0.08838834764831845f;           // 1/sqrt(128)

    float sc[12];
    int   jd[12];
    int   cnt = 0;

    const int e1 = (i + 1 < 4) ? (i + 1) : 4;
    int s2 = i - 7; if (s2 < e1) s2 = e1;

    // sink tokens
    for (int j = 0; j < e1; ++j) {
        const float4 kv = *(const float4*)(Kp + (size_t)(b * AS_S + j) * 1024 + kvoff);
        float p = qv.x * kv.x + qv.y * kv.y + qv.z * kv.z + qv.w * kv.w;
#pragma unroll
        for (int o = 16; o > 0; o >>= 1) p += __shfl_xor(p, o, 32);
        sc[cnt] = p * scale; jd[cnt] = j; ++cnt;
    }
    // local window
    for (int j = s2; j <= i; ++j) {
        const float4 kv = *(const float4*)(Kp + (size_t)(b * AS_S + j) * 1024 + kvoff);
        float p = qv.x * kv.x + qv.y * kv.y + qv.z * kv.z + qv.w * kv.w;
#pragma unroll
        for (int o = 16; o > 0; o >>= 1) p += __shfl_xor(p, o, 32);
        sc[cnt] = p * scale; jd[cnt] = j; ++cnt;
    }

    // softmax over <=12 scores (identical across lanes after reduction)
    float m = -3.0e38f;
    for (int q = 0; q < cnt; ++q) m = fmaxf(m, sc[q]);
    float sum = 0.f;
    for (int q = 0; q < cnt; ++q) { sc[q] = __expf(sc[q] - m); sum += sc[q]; }
    const float inv = 1.f / sum;

    float4 o4 = make_float4(0.f, 0.f, 0.f, 0.f);
    for (int q = 0; q < cnt; ++q) {
        const float  wt = sc[q] * inv;
        const float4 vv = *(const float4*)(Vp + (size_t)(b * AS_S + jd[q]) * 1024 + kvoff);
        o4.x += wt * vv.x; o4.y += wt * vv.y; o4.z += wt * vv.z; o4.w += wt * vv.w;
    }
    *(float4*)(O + qoff) = o4;
}

// ---------------------------------------------------------------------------
// Orchestration
// ---------------------------------------------------------------------------
extern "C" void kernel_launch(void* const* d_in, const int* in_sizes, int n_in,
                              void* d_out, int out_size, void* d_ws, size_t ws_size,
                              hipStream_t stream) {
    const float* x  = (const float*)d_in[0];   // [2,2048,2048]
    const float* Wq = (const float*)d_in[1];   // [2048,2048]
    const float* Wk = (const float*)d_in[2];   // [1024,2048]
    const float* Wv = (const float*)d_in[3];   // [1024,2048]
    const float* Wo = (const float*)d_in[4];   // [2048,2048]

    const int NX  = 2 * 2048 * 2048;  // 8,388,608
    const int NWQ = 2048 * 2048;      // 4,194,304
    const int NWK = 1024 * 2048;      // 2,097,152

    char* ws = (char*)d_ws;
    size_t off = 0;
    __bf16* xb    = (__bf16*)(ws + off); off += (size_t)NX  * 2;
    __bf16* wqb   = (__bf16*)(ws + off); off += (size_t)NWQ * 2;
    __bf16* wkb   = (__bf16*)(ws + off); off += (size_t)NWK * 2;
    __bf16* wvb   = (__bf16*)(ws + off); off += (size_t)NWK * 2;
    __bf16* wob   = (__bf16*)(ws + off); off += (size_t)NWQ * 2;
    float*  Qf    = (float*) (ws + off); off += (size_t)NX  * 4;
    float*  Kf    = (float*) (ws + off); off += (size_t)NX / 2 * 4;
    float*  Vf    = (float*) (ws + off); off += (size_t)NX / 2 * 4;
    float*  attnf = (float*) (ws + off); off += (size_t)NX  * 4;
    __bf16* attnb = (__bf16*)(ws + off); off += (size_t)NX  * 2;

    // 1) downconvert activations + weights to bf16
    cvt_f32_bf16<<<(NX  + 255) / 256, 256, 0, stream>>>(x,  xb,  NX);
    cvt_f32_bf16<<<(NWQ + 255) / 256, 256, 0, stream>>>(Wq, wqb, NWQ);
    cvt_f32_bf16<<<(NWK + 255) / 256, 256, 0, stream>>>(Wk, wkb, NWK);
    cvt_f32_bf16<<<(NWK + 255) / 256, 256, 0, stream>>>(Wv, wvb, NWK);
    cvt_f32_bf16<<<(NWQ + 255) / 256, 256, 0, stream>>>(Wo, wob, NWQ);

    // 2) projections: [4096,2048] x [N,2048]^T
    const int M = 4096, Kdim = 2048;
    gemm_bf16_wmma<<<dim3(2048 / BN, M / BM), 256, 0, stream>>>(xb, wqb, Qf, M, 2048, Kdim);
    gemm_bf16_wmma<<<dim3(1024 / BN, M / BM), 256, 0, stream>>>(xb, wkb, Kf, M, 1024, Kdim);
    gemm_bf16_wmma<<<dim3(1024 / BN, M / BM), 256, 0, stream>>>(xb, wvb, Vf, M, 1024, Kdim);

    // 3) sparse streaming attention: 65536 waves = 8192 blocks of 8 waves
    attn_stream<<<(2 * 16 * 2048 * 32) / 256, 256, 0, stream>>>(Qf, Kf, Vf, attnf);

    // 4) output projection
    cvt_f32_bf16<<<(NX + 255) / 256, 256, 0, stream>>>(attnf, attnb, NX);
    gemm_bf16_wmma<<<dim3(2048 / BN, M / BM), 256, 0, stream>>>(attnb, wob, (float*)d_out, M, 2048, Kdim);
}
